// VQVAEEncoder_73933567033596
// MI455X (gfx1250) — matrix operand order repaired and checked by reading, not compiled, not run on hardware
//
#include <hip/hip_runtime.h>

// ---------------------------------------------------------------------------
// VQ-VAE encoder for MI455X (gfx1250, wave32, WMMA, async global->LDS).
//
//   conv1 (direct VALU, C=3)            -> act1 [4,128,128,128] f16 channels-last
//   conv2 (WMMA implicit GEMM, K=2048)  -> act2 [4, 64, 64,128] f16 channels-last
//   conv3 (WMMA implicit GEMM, K=2048)  -> act3 [4, 32, 32,128] f16 channels-last
//   conv4 (WMMA implicit GEMM, K=1152)  -> z    [4096,32]       f32
//   pack codebook -> f16 hi/lo split + exact fp32 ||e||^2
//   vq_dist: 8 waves/WG share LDS-staged 16-code tiles (async b128 copies,
//            double buffered); 3x v_wmma_f32_16x16x32_f16 per tile per wave
//            (split-f16 ~fp32 dot); hierarchical argmin with index tie-break
//   finalize: reduce K-slices per row, gather z_q
// ---------------------------------------------------------------------------

typedef __attribute__((ext_vector_type(16))) _Float16 v16h;
typedef __attribute__((ext_vector_type(8)))  float    v8f;
typedef __attribute__((ext_vector_type(4)))  float    f4;
typedef __attribute__((ext_vector_type(4)))  int      i32x4;

#if __has_builtin(__builtin_amdgcn_global_load_async_to_lds_b128)
#define HAVE_ASYNC_LDS 1
typedef __attribute__((address_space(1))) i32x4 as1_i32x4;
typedef __attribute__((address_space(3))) i32x4 as3_i32x4;
#endif

__device__ __forceinline__ void wait_async0() {
#if __has_builtin(__builtin_amdgcn_s_wait_asynccnt)
  __builtin_amdgcn_s_wait_asynccnt(0);
#else
  asm volatile("s_wait_asynccnt 0x0" ::: "memory");
#endif
}

// A-matrix (16-bit, 16x32) per-lane slot -> K index (ISA 7.12.2 table)
__device__ __forceinline__ int a_slot_k(int i, int h) {
  return (i < 8) ? (8 * h + i) : (16 + 8 * h + (i - 8));
}

// ---------------------------------------------------------------------------
// conv1: NCHW f32 input, 4x4 s2 p1, C=3 -> channels-last f16 out + ReLU
// ---------------------------------------------------------------------------
__global__ void conv1_kernel(const float* __restrict__ x,
                             const float* __restrict__ w1,
                             const float* __restrict__ b1,
                             _Float16* __restrict__ act1) {
  int oc  = threadIdx.x;            // 0..127
  int pix = blockIdx.x;             // 0..65535  (n, oh, ow) with ow fastest
  int ow = pix & 127;
  int oh = (pix >> 7) & 127;
  int n  = pix >> 14;
  float s = b1[oc];
  for (int c = 0; c < 3; ++c)
    for (int kh = 0; kh < 4; ++kh) {
      int ih = oh * 2 - 1 + kh;
      if ((unsigned)ih >= 256u) continue;
      for (int kw = 0; kw < 4; ++kw) {
        int iw = ow * 2 - 1 + kw;
        if ((unsigned)iw >= 256u) continue;
        s += x[(((size_t)n * 3 + c) * 256 + ih) * 256 + iw] *
             w1[(((size_t)oc * 3 + c) * 4 + kh) * 4 + kw];
      }
    }
  s = s > 0.f ? s : 0.f;
  act1[(size_t)pix * 128 + oc] = (_Float16)s;
}

// ---------------------------------------------------------------------------
// Weight pack: OIHW f32 -> A-matrix lane order f16.
// im2col K ordering: K = (kh*KW + kw)*Cin + c.
// pw layout: [octile][chunk][lane 0..31][slot 0..15]
// ---------------------------------------------------------------------------
__global__ void pack_w_kernel(const float* __restrict__ w,
                              _Float16* __restrict__ pw,
                              int OC, int Cin, int KH, int KW) {
  int gid = blockIdx.x * blockDim.x + threadIdx.x;
  int chunks = (KH * KW * Cin) / 32;
  int total  = (OC / 16) * chunks * 32;
  if (gid >= total) return;
  int lane   = gid & 31;
  int chunk  = (gid >> 5) % chunks;
  int octile = (gid >> 5) / chunks;
  int oc = octile * 16 + (lane & 15);
  int h  = lane >> 4;
#pragma unroll
  for (int i = 0; i < 16; ++i) {
    int K   = chunk * 32 + a_slot_k(i, h);
    int tap = K / Cin;
    int c   = K % Cin;
    int kh  = tap / KW;
    int kw  = tap % KW;
    float v = w[(((size_t)oc * Cin + c) * KH + kh) * KW + kw];
    pw[(size_t)gid * 16 + i] = (_Float16)v;
  }
}

// ---------------------------------------------------------------------------
// Codebook pack: row-major f32 -> row-major f16 hi + f16 lo residual + ||e||^2
// ---------------------------------------------------------------------------
__global__ void pack_cb_kernel(const float* __restrict__ cb,
                               _Float16* __restrict__ cbh,
                               _Float16* __restrict__ cbl,
                               float* __restrict__ norme, int Kc) {
  int code = blockIdx.x * blockDim.x + threadIdx.x;
  if (code >= Kc) return;
  const float* r = cb + (size_t)code * 32;
  float s = 0.f;
#pragma unroll
  for (int k = 0; k < 32; ++k) {
    float f = r[k];
    s += f * f;
    _Float16 hi = (_Float16)f;
    cbh[(size_t)code * 32 + k] = hi;
    cbl[(size_t)code * 32 + k] = (_Float16)(f - (float)hi);
  }
  norme[code] = s;
}

// ---------------------------------------------------------------------------
// Implicit-GEMM conv via v_wmma_f32_16x16x32_f16, all dims compile-time.
// One wave per (16 OC x 16 pixels) tile; K consumed 32/step, fully unrolled.
// ---------------------------------------------------------------------------
template <int Cin, int IH, int IW, int OC, int OH, int OW,
          int KH, int KW, int STRIDE, int PAD, bool RELU, bool OUT_F16>
__global__ __launch_bounds__(32)
void conv_wmma_kernel(const _Float16* __restrict__ act_in,  // [N][IH][IW][Cin]
                      const _Float16* __restrict__ pw,
                      const float* __restrict__ bias,
                      _Float16* __restrict__ act_out,       // if OUT_F16
                      float* __restrict__ f32_out) {        // else
  constexpr int CHUNKS  = (KH * KW * Cin) / 32;
  constexpr int CCHUNKS = Cin / 32;
  int lane   = threadIdx.x;
  int ptile  = blockIdx.x;
  int octile = blockIdx.y;
  int h   = lane >> 4;
  int col = lane & 15;
  int pix = ptile * 16 + col;           // 16 consecutive ow (OW % 16 == 0)
  int ow = pix % OW;
  int oh = (pix / OW) % OH;
  int n  = pix / (OW * OH);
  const _Float16* wbase = pw + (size_t)octile * CHUNKS * 512;

  v8f acc = {};
#pragma unroll
  for (int ch = 0; ch < CHUNKS; ++ch) {
    v16h a = *(const v16h*)(wbase + (size_t)ch * 512 + lane * 16);
    int tap = ch / CCHUNKS;
    int c0  = (ch % CCHUNKS) * 32;
    int kh  = tap / KW;
    int kw  = tap % KW;
    int ih  = oh * STRIDE - PAD + kh;
    int iw  = ow * STRIDE - PAD + kw;
    v16h b;
#pragma unroll
    for (int i = 0; i < 16; ++i) b[i] = (_Float16)0.0f;
    if ((unsigned)ih < (unsigned)IH && (unsigned)iw < (unsigned)IW) {
      const _Float16* bp =
          act_in + (((size_t)(n * IH + ih) * IW + iw) * Cin + c0 + h * 16);
      typedef __attribute__((ext_vector_type(8))) _Float16 v8h_;
      v8h_ b0 = *(const v8h_*)(bp - h * 8);
      v8h_ b1 = *(const v8h_*)(bp - h * 8 + 16);
#pragma unroll
      for (int i = 0; i < 8; ++i) { b[i] = b0[i]; b[8 + i] = b1[i]; }
    }
    acc = __builtin_amdgcn_wmma_f32_16x16x32_f16(false, a, false, b,
                                                 (short)0, acc, false, false);
  }
  // D layout: lane holds column N=col; VGPR r -> row M = r + 8*h
#pragma unroll
  for (int r = 0; r < 8; ++r) {
    int oc = octile * 16 + r + 8 * h;
    float v = acc[r] + bias[oc];
    if (RELU) v = v > 0.f ? v : 0.f;
    size_t o = (size_t)pix * OC + oc;
    if (OUT_F16) act_out[o] = (_Float16)v;
    else         f32_out[o] = v;
  }
}

// ---------------------------------------------------------------------------
// Distance + partial argmin.
// Workgroup = 256 threads = 8 waves; each wave owns one 16-row z tile, all
// waves share each 16-code tile staged in LDS (2 KB: f16-hi 1 KB + f16-lo
// 1 KB), double buffered, copied with global_load_async_to_lds_b128 by wave 0
// (ASYNCcnt + s_wait_asynccnt + workgroup barrier). This cuts codebook L2
// traffic 8x vs per-wave streaming.
// ---------------------------------------------------------------------------
#define KSLICES 16
#define TILES_PER_SLICE 256   // 65536 codes / 16 per tile / KSLICES
#define WAVES_PER_WG 8

__global__ __launch_bounds__(256)
void vq_dist_kernel(const float* __restrict__ z,        // [4096][32]
                    const _Float16* __restrict__ cbh,   // [65536][32]
                    const _Float16* __restrict__ cbl,
                    const float* __restrict__ norme,    // [65536]
                    float* __restrict__ pminv,          // [4096][KSLICES]
                    int* __restrict__ pmini) {
  __shared__ __align__(64) _Float16 lds[2][1024];  // [buf][hi 512 | lo 512]

  int tid    = threadIdx.x;
  int wave   = tid >> 5;
  int lane   = tid & 31;
  int zgroup = blockIdx.x;   // 0..31
  int kslice = blockIdx.y;   // 0..15
  int ztile  = zgroup * WAVES_PER_WG + wave;
  int h   = lane >> 4;
  int col = lane & 15;

  // Build A operands (z rows, split f16 hi/lo) in contiguous k order (k = 16h + i).
  const float* zr = z + ((size_t)ztile * 16 + col) * 32;
  v16h zh, zl;
#pragma unroll
  for (int i = 0; i < 16; ++i) {
    float f = zr[16 * h + i];
    _Float16 hi = (_Float16)f;
    zh[i] = hi;
    zl[i] = (_Float16)(f - (float)hi);
  }

  float minv[8];
  int   mini[8];
#pragma unroll
  for (int r = 0; r < 8; ++r) { minv[r] = 3.4e38f; mini[r] = 0; }

  const int tile0 = kslice * TILES_PER_SLICE;

  // Stage code tile T (2 KB: 1 KB hi + 1 KB lo, contiguous) into lds[buf].
  auto stage = [&](int T, int buf) {
#ifdef HAVE_ASYNC_LDS
    if (wave == 0) {
      char* lbase = (char*)&lds[buf][0];
#pragma unroll
      for (int s = 0; s < 2; ++s) {
        size_t go = (size_t)T * 1024 + s * 512 + (size_t)lane * 16;
        size_t lo = (size_t)s * 512 + (size_t)lane * 16;
        __builtin_amdgcn_global_load_async_to_lds_b128(
            (as1_i32x4*)(uintptr_t)((const char*)cbh + go),
            (as3_i32x4*)(lbase + lo), 0, 0);
        __builtin_amdgcn_global_load_async_to_lds_b128(
            (as1_i32x4*)(uintptr_t)((const char*)cbl + go),
            (as3_i32x4*)(lbase + 1024 + lo), 0, 0);
      }
      // L2 lookahead for upcoming tiles
      __builtin_prefetch((const char*)cbh + (size_t)T * 1024 + 8192, 0, 3);
      __builtin_prefetch((const char*)cbl + (size_t)T * 1024 + 8192, 0, 3);
    }
#else
    if (tid < 128) {  // 128 threads x 16 B = 2 KB
      int s = tid >> 5;              // 0..3 segment
      size_t go = (size_t)T * 1024 + (s & 1) * 512 + (size_t)lane * 16;
      size_t lo = (s & 1) * 512 + (size_t)lane * 16;
      const char* src = ((s < 2) ? (const char*)cbh : (const char*)cbl) + go;
      char* dst = (char*)&lds[buf][0] + ((s < 2) ? 0 : 1024) + lo;
      *(f4*)dst = *(const f4*)src;
    }
#endif
  };

  stage(tile0, 0);
#ifdef HAVE_ASYNC_LDS
  if (wave == 0) wait_async0();
#endif
  __syncthreads();

  for (int t = 0; t < TILES_PER_SLICE; ++t) {
    int buf = t & 1;
    if (t + 1 < TILES_PER_SLICE) stage(tile0 + t + 1, buf ^ 1);

    int T = tile0 + t;
    int code = T * 16 + col;                       // this lane's B column
    const _Float16* lb = &lds[buf][0];
    size_t lo = (size_t)col * 32 + h * 16;         // 16 contiguous halves
    v16h bh = *(const v16h*)(lb + lo);
    v16h bl = *(const v16h*)(lb + 512 + lo);

    v8f c = {};
    c = __builtin_amdgcn_wmma_f32_16x16x32_f16(false, zh, false, bl,
                                               (short)0, c, false, false);
    c = __builtin_amdgcn_wmma_f32_16x16x32_f16(false, zl, false, bh,
                                               (short)0, c, false, false);
    c = __builtin_amdgcn_wmma_f32_16x16x32_f16(false, zh, false, bh,
                                               (short)0, c, false, false);

    float nrm = norme[code];
#pragma unroll
    for (int r = 0; r < 8; ++r) {
      float d = nrm - 2.0f * c[r];
      if (d < minv[r] || (d == minv[r] && code < mini[r])) {
        minv[r] = d;
        mini[r] = code;
      }
    }

#ifdef HAVE_ASYNC_LDS
    if (wave == 0) wait_async0();
#endif
    __syncthreads();
  }

  // Reduce across the 16 lanes of each half (rows M = r + 8h).
#pragma unroll
  for (int r = 0; r < 8; ++r) {
    float v = minv[r];
    int   i = mini[r];
    for (int m = 8; m >= 1; m >>= 1) {
      float ov = __shfl_xor(v, m, 16);
      int   oi = __shfl_xor(i, m, 16);
      if (ov < v || (ov == v && oi < i)) { v = ov; i = oi; }
    }
    if (col == 0) {
      int grow = ztile * 16 + r + 8 * h;
      pminv[(size_t)grow * KSLICES + kslice] = v;
      pmini[(size_t)grow * KSLICES + kslice] = i;
    }
  }
}

// ---------------------------------------------------------------------------
// Final reduce over K-slices (lowest-index tie-break == jnp.argmin) + gather.
// ---------------------------------------------------------------------------
__global__ void finalize_kernel(const float* __restrict__ pminv,
                                const int* __restrict__ pmini,
                                const float* __restrict__ cb,
                                int* __restrict__ idx_out,
                                float* __restrict__ zq_out) {
  int row = blockIdx.x * blockDim.x + threadIdx.x;
  if (row >= 4096) return;
  float bv = 3.4e38f;
  int   bi = 0;
  for (int s = 0; s < KSLICES; ++s) {
    float v = pminv[(size_t)row * KSLICES + s];
    int   i = pmini[(size_t)row * KSLICES + s];
    if (v < bv || (v == bv && i < bi)) { bv = v; bi = i; }
  }
  idx_out[row] = bi;
  const float* r = cb + (size_t)bi * 32;
#pragma unroll
  for (int d = 0; d < 32; ++d) zq_out[(size_t)row * 32 + d] = r[d];
}

// ---------------------------------------------------------------------------
extern "C" void kernel_launch(void* const* d_in, const int* in_sizes, int n_in,
                              void* d_out, int out_size, void* d_ws,
                              size_t ws_size, hipStream_t stream) {
  (void)in_sizes; (void)n_in; (void)out_size; (void)ws_size;
  const float* x  = (const float*)d_in[0];
  const float* w1 = (const float*)d_in[1];
  const float* b1 = (const float*)d_in[2];
  const float* w2 = (const float*)d_in[3];
  const float* b2 = (const float*)d_in[4];
  const float* w3 = (const float*)d_in[5];
  const float* b3 = (const float*)d_in[6];
  const float* w4 = (const float*)d_in[7];
  const float* b4 = (const float*)d_in[8];
  const float* cb = (const float*)d_in[9];

  char* p = (char*)d_ws;
  auto alloc = [&](size_t bytes) -> void* {
    char* r = p;
    p += (bytes + 255) & ~(size_t)255;
    return (void*)r;
  };
  _Float16* act1 = (_Float16*)alloc((size_t)8388608 * 2);  // 4*128*128*128
  _Float16* act2 = (_Float16*)alloc((size_t)2097152 * 2);  // 4*64*64*128
  _Float16* act3 = (_Float16*)alloc((size_t)524288 * 2);   // 4*32*32*128
  float*    zbuf = (float*)alloc((size_t)131072 * 4);      // 4096*32
  _Float16* pw2  = (_Float16*)alloc((size_t)262144 * 2);   // 128*2048
  _Float16* pw3  = (_Float16*)alloc((size_t)262144 * 2);
  _Float16* pw4  = (_Float16*)alloc((size_t)36864 * 2);    // 32*1152
  _Float16* cbh  = (_Float16*)alloc((size_t)2097152 * 2);  // 65536*32
  _Float16* cbl  = (_Float16*)alloc((size_t)2097152 * 2);
  float*    nrm  = (float*)alloc((size_t)65536 * 4);
  float*    pmv  = (float*)alloc((size_t)65536 * 4);       // 4096*16
  int*      pmi  = (int*)alloc((size_t)65536 * 4);

  // One-time packs.
  pack_cb_kernel<<<65536 / 256, 256, 0, stream>>>(cb, cbh, cbl, nrm, 65536);
  pack_w_kernel<<<(8 * 64 * 32 + 255) / 256, 256, 0, stream>>>(w2, pw2, 128, 128, 4, 4);
  pack_w_kernel<<<(8 * 64 * 32 + 255) / 256, 256, 0, stream>>>(w3, pw3, 128, 128, 4, 4);
  pack_w_kernel<<<(2 * 36 * 32 + 255) / 256, 256, 0, stream>>>(w4, pw4, 32, 128, 3, 3);

  // Encoder.
  conv1_kernel<<<65536, 128, 0, stream>>>(x, w1, b1, act1);
  conv_wmma_kernel<128, 128, 128, 128, 64, 64, 4, 4, 2, 1, true, true>
      <<<dim3(16384 / 16, 8), 32, 0, stream>>>(act1, pw2, b2, act2, nullptr);
  conv_wmma_kernel<128, 64, 64, 128, 32, 32, 4, 4, 2, 1, true, true>
      <<<dim3(4096 / 16, 8), 32, 0, stream>>>(act2, pw3, b3, act3, nullptr);
  conv_wmma_kernel<128, 32, 32, 32, 32, 32, 3, 3, 1, 1, false, false>
      <<<dim3(4096 / 16, 2), 32, 0, stream>>>(act3, pw4, b4, nullptr, zbuf);

  // Vector quantization.
  vq_dist_kernel<<<dim3(32, KSLICES), 256, 0, stream>>>(zbuf, cbh, cbl, nrm,
                                                        pmv, pmi);
  finalize_kernel<<<(4096 + 255) / 256, 256, 0, stream>>>(
      pmv, pmi, cb, (int*)d_out, ((float*)d_out) + 4096);
}
